// AttentionMixer_85727547228251
// MI455X (gfx1250) — compile-verified, matrix operand order checked by reading, not run on hardware
//
#include <hip/hip_runtime.h>
#include <hip/hip_bf16.h>

typedef __attribute__((ext_vector_type(16))) _Float16 v16h;
typedef __attribute__((ext_vector_type(4)))  _Float16 v4h;
typedef __attribute__((ext_vector_type(8)))  float    v8f;

#define DMODEL 512
#define NROWS  32768          // 16 * 256 * 8
#define SEQLEN 128
#define HEADS  8
#define EDIM   64

// Involutive row permutation: r = b<<11 | hi<<7 | lo<<3 | p  ->  swap hi/lo 4-bit fields
__device__ __forceinline__ int permrow(int r) {
    int b  = r >> 11;
    int hi = (r >> 7) & 15;
    int lo = (r >> 3) & 15;
    int p  = r & 7;
    return (b << 11) | (lo << 7) | (hi << 3) | p;
}

// Load a 16x32 (MxK) f16 WMMA A/B fragment for this lane from a row pointer.
// Lane layout (ISA 7.12.2): lane holds K = {k0..k0+7} U {k0+16..k0+23}
__device__ __forceinline__ v16h load_frag_f16(const _Float16* __restrict__ p, int k0) {
    v16h a;
#pragma unroll
    for (int i = 0; i < 8; ++i) {
        a[i]     = p[k0 + i];
        a[i + 8] = p[k0 + 16 + i];
    }
    return a;
}

__device__ __forceinline__ v8f wmma_f16(v16h a, v16h b, v8f c) {
    return __builtin_amdgcn_wmma_f32_16x16x32_f16(false, a, false, b, (short)0, c, false, false);
}

// ---------------------------------------------------------------------------
// Elementwise f32 -> f16 conversion (4 elems/thread, b128 load -> b64 store)
// ---------------------------------------------------------------------------
__global__ __launch_bounds__(256) void cvt_f32_f16_kernel(const float* __restrict__ s,
                                                          _Float16* __restrict__ d,
                                                          int n4) {
    int i = blockIdx.x * 256 + threadIdx.x;
    if (i < n4) {
        float4 f = ((const float4*)s)[i];
        v4h h;
        h[0] = (_Float16)f.x;
        h[1] = (_Float16)f.y;
        h[2] = (_Float16)f.z;
        h[3] = (_Float16)f.w;
        ((v4h*)d)[i] = h;
    }
}

// ---------------------------------------------------------------------------
// Y[M,512] = X[M,512] @ W[512,512]^T + bias   (nn.Linear semantics)
// X, W are f16. OUTF32PERM: 0 -> Y f16 direct; 1 -> Y f32, rows at permrow(r)
// grid = (M/16, 8), block = 128 (4 waves); wave w owns cols blockIdx.y*64+w*16
// ---------------------------------------------------------------------------
template <int OUTF32PERM>
__global__ __launch_bounds__(128) void gemm512_kernel(const _Float16* __restrict__ X,
                                                      const _Float16* __restrict__ W,
                                                      const float* __restrict__ bias,
                                                      void* __restrict__ Yv) {
    const int lane = threadIdx.x & 31;
    const int wave = threadIdx.x >> 5;
    const int mt   = blockIdx.x;                 // 16-row tile index
    const int lrow = lane & 15;
    const int kg   = (lane >> 4) * 8;            // 0 or 8: K sub-group / C row offset
    const int ncol = blockIdx.y * 64 + wave * 16 + lrow;

    const _Float16* __restrict__ xrow = X + (size_t)(mt * 16 + lrow) * DMODEL;
    const _Float16* __restrict__ wrow = W + (size_t)ncol * DMODEL;   // B col n == W row n

    v8f acc = {0.f, 0.f, 0.f, 0.f, 0.f, 0.f, 0.f, 0.f};
#pragma unroll
    for (int kt = 0; kt < DMODEL / 32; ++kt) {
        if (kt + 2 < DMODEL / 32)
            __builtin_prefetch(xrow + (kt + 2) * 32, 0, 3);   // global_prefetch_b8
        v16h a = load_frag_f16(xrow, kt * 32 + kg);
        v16h b = load_frag_f16(wrow, kt * 32 + kg);
        acc = wmma_f16(a, b, acc);
    }

    const float bv = bias[ncol];
    const int r0 = mt * 16 + kg;                 // C/D rows: (lane>>4)*8 + i
    if (OUTF32PERM) {
        float* __restrict__ Y = (float*)Yv;
#pragma unroll
        for (int i = 0; i < 8; ++i) {
            int r = permrow(r0 + i);
            Y[(size_t)r * DMODEL + ncol] = acc[i] + bv;
        }
    } else {
        _Float16* __restrict__ Y = (_Float16*)Yv;
#pragma unroll
        for (int i = 0; i < 8; ++i)
            Y[(size_t)(r0 + i) * DMODEL + ncol] = (_Float16)(acc[i] + bv);
    }
}

// ---------------------------------------------------------------------------
// One workgroup per (seq, head): softmax(QK^T / sqrt(E)) @ V
// Q,K,V,O: f16 [32768, 512], head h occupies cols [h*64, h*64+64)
// block = 256 (8 waves), dynamic LDS = 128*128*4 = 64KB
// PERM: output rows pass through permrow()
// ---------------------------------------------------------------------------
template <int PERM>
__global__ __launch_bounds__(256) void attn_kernel(const _Float16* __restrict__ Q,
                                                   const _Float16* __restrict__ K,
                                                   const _Float16* __restrict__ V,
                                                   _Float16* __restrict__ O) {
    extern __shared__ float S[];                 // [128][128] scores, reused for f16 A
    const int tid  = threadIdx.x;
    const int lane = tid & 31;
    const int wave = tid >> 5;
    const int seq  = blockIdx.x >> 3;
    const int head = blockIdx.x & 7;
    const int lrow = lane & 15;
    const int kg   = (lane >> 4) * 8;

    // ---- Phase 1: S = Q K^T (per-wave 16-row stripe, all 8 col tiles) ----
    const _Float16* __restrict__ qp =
        Q + (size_t)(seq * SEQLEN + wave * 16 + lrow) * DMODEL + head * EDIM;
    v16h aq0 = load_frag_f16(qp, kg);
    v16h aq1 = load_frag_f16(qp, 32 + kg);
#pragma unroll
    for (int ct = 0; ct < 8; ++ct) {
        const _Float16* __restrict__ kp =
            K + (size_t)(seq * SEQLEN + ct * 16 + lrow) * DMODEL + head * EDIM;
        v16h bk0 = load_frag_f16(kp, kg);
        v16h bk1 = load_frag_f16(kp, 32 + kg);
        v8f s = {0.f, 0.f, 0.f, 0.f, 0.f, 0.f, 0.f, 0.f};
        s = wmma_f16(aq0, bk0, s);
        s = wmma_f16(aq1, bk1, s);
        const int col = ct * 16 + lrow;
        const int rb  = wave * 16 + kg;
#pragma unroll
        for (int i = 0; i < 8; ++i) S[(rb + i) * SEQLEN + col] = s[i];
    }
    __syncthreads();

    // ---- Softmax: one thread per row; overwrite row in place with f16 A ----
    if (tid < SEQLEN) {
        const float scale = 0.125f;              // 1/sqrt(64)
        float* Sr = S + tid * SEQLEN;
        float m = -3.0e38f;
        for (int c = 0; c < SEQLEN; ++c) m = fmaxf(m, Sr[c]);
        m *= scale;
        _Float16* Ar = (_Float16*)Sr;            // half c clobbers float c>>1 <= c : safe
        float sum = 0.f;
        for (int c = 0; c < SEQLEN; ++c) {
            float e = __expf(Sr[c] * scale - m);
            sum += e;
            Ar[c] = (_Float16)e;
        }
        float inv = 1.0f / sum;
        for (int c = 0; c < SEQLEN; ++c) Ar[c] = (_Float16)((float)Ar[c] * inv);
    }
    __syncthreads();

    // ---- Phase 2: O = A @ V (A from LDS, row stride 256 halves) ----
    const _Float16* __restrict__ Arow =
        (const _Float16*)S + (wave * 16 + lrow) * (2 * SEQLEN);
    v16h afrag[4];
#pragma unroll
    for (int kt = 0; kt < 4; ++kt) afrag[kt] = load_frag_f16(Arow, kt * 32 + kg);

#pragma unroll
    for (int ct = 0; ct < 4; ++ct) {
        v8f o = {0.f, 0.f, 0.f, 0.f, 0.f, 0.f, 0.f, 0.f};
#pragma unroll
        for (int kt = 0; kt < 4; ++kt) {
            // B col n = V column (head*64 + ct*16 + n), K rows kt*32 + {kg..}:
            const _Float16* __restrict__ vp =
                V + (size_t)(seq * SEQLEN + kt * 32) * DMODEL + head * EDIM + ct * 16 + lrow;
            v16h bv;
#pragma unroll
            for (int i = 0; i < 8; ++i) {
                bv[i]     = vp[(size_t)(kg + i) * DMODEL];
                bv[i + 8] = vp[(size_t)(kg + 16 + i) * DMODEL];
            }
            o = wmma_f16(afrag[kt], bv, o);
        }
        const int col = head * EDIM + ct * 16 + lrow;
        const int rb  = wave * 16 + kg;
#pragma unroll
        for (int i = 0; i < 8; ++i) {
            int r = seq * SEQLEN + rb + i;
            if (PERM) r = permrow(r);
            O[(size_t)r * DMODEL + col] = (_Float16)o[i];
        }
    }
}

extern "C" void kernel_launch(void* const* d_in, const int* in_sizes, int n_in,
                              void* d_out, int out_size, void* d_ws, size_t ws_size,
                              hipStream_t stream) {
    const float* x   = (const float*)d_in[0];
    const float* wsrc[7] = {
        (const float*)d_in[1],  (const float*)d_in[3],  (const float*)d_in[5],
        (const float*)d_in[7],  (const float*)d_in[9],  (const float*)d_in[11],
        (const float*)d_in[13]
    };
    const float* bsrc[7] = {
        (const float*)d_in[2],  (const float*)d_in[4],  (const float*)d_in[6],
        (const float*)d_in[8],  (const float*)d_in[10], (const float*)d_in[12],
        (const float*)d_in[14]
    };

    const size_t NELEM = (size_t)NROWS * DMODEL;       // 16.7M f16 elems = 32MB
    const size_t WELEM = (size_t)DMODEL * DMODEL;      // 256K f16 elems = 512KB

    _Float16* xh = (_Float16*)d_ws;
    _Float16* q  = xh + NELEM;
    _Float16* k  = q + NELEM;
    _Float16* v  = k + NELEM;
    _Float16* h  = v + NELEM;
    _Float16* wh[7];
    wh[0] = h + NELEM;
    for (int i = 1; i < 7; ++i) wh[i] = wh[i - 1] + WELEM;

    // ---- f32 -> f16 pre-pass (x + 7 weight matrices) ----
    {
        int n4 = (int)(NELEM / 4);
        cvt_f32_f16_kernel<<<(n4 + 255) / 256, 256, 0, stream>>>(x, xh, n4);
        int w4 = (int)(WELEM / 4);
        for (int i = 0; i < 7; ++i)
            cvt_f32_f16_kernel<<<(w4 + 255) / 256, 256, 0, stream>>>(wsrc[i], wh[i], w4);
    }

    dim3 gg(NROWS / 16, DMODEL / 64), gb(128);
    dim3 ag(256 * HEADS), ab(256);
    const size_t lds = SEQLEN * SEQLEN * sizeof(float);   // 64 KB

    // QKV projections of layer 1 (attention-1 row order is already contiguous)
    gemm512_kernel<0><<<gg, gb, 0, stream>>>(xh, wh[0], bsrc[0], q);
    gemm512_kernel<0><<<gg, gb, 0, stream>>>(xh, wh[1], bsrc[1], k);
    gemm512_kernel<0><<<gg, gb, 0, stream>>>(xh, wh[2], bsrc[2], v);

    // attention 1, output rows written through ng<->cg swap so layer 2 reads contiguously
    attn_kernel<1><<<ag, ab, lds, stream>>>(q, k, v, h);

    // QKV projections of layer 2
    gemm512_kernel<0><<<gg, gb, 0, stream>>>(h, wh[3], bsrc[3], q);
    gemm512_kernel<0><<<gg, gb, 0, stream>>>(h, wh[4], bsrc[4], k);
    gemm512_kernel<0><<<gg, gb, 0, stream>>>(h, wh[5], bsrc[5], v);

    // attention 2, plain row order
    attn_kernel<0><<<ag, ab, lds, stream>>>(q, k, v, h);

    // output projection, f32, rows permuted back to [B, ch, num_p, dim] order
    gemm512_kernel<1><<<gg, gb, 0, stream>>>(h, wh[6], bsrc[6], d_out);
}
